// EdgeAwareCrossTokenRefine_72791105733032
// MI455X (gfx1250) — matrix operand order repaired and checked by reading, not compiled, not run on hardware
//
#include <hip/hip_runtime.h>
#include <hip/hip_bf16.h>
#include <math.h>

// ---------------------------------------------------------------------------
// EdgeAwareCrossTokenRefine for MI455X (gfx1250, wave32, WMMA bf16)
// B=4, N=2048, C=512, H=4, hd=128.
// out = x + gate*(1+0.5*edge)*softmax(Qn Kn^T/sqrt(hd)) V
// ---------------------------------------------------------------------------

#define Bsz 4
#define Ntok 2048
#define Cdim 512
#define Hh 4
#define HD 128
#define BN 64                      // keys/values per inner tile
#define NIT (Ntok / BN)            // 32 column iterations
#define NTOK_TOTAL (Bsz * Ntok)    // 8192

typedef __attribute__((ext_vector_type(16))) __bf16 v16bf;
typedef __attribute__((ext_vector_type(8)))  float  v8f;

union FragU { uint4 u[2]; v16bf v; };

// ----- gfx1250 async copy to LDS (ASYNCcnt path), with safe fallback -------
#if defined(__has_builtin)
#if __has_builtin(__builtin_amdgcn_global_load_async_to_lds_b128)
#define USE_ASYNC 1
#endif
#endif
#ifndef USE_ASYNC
#define USE_ASYNC 0
#endif

typedef int v4i __attribute__((vector_size(16)));
typedef __attribute__((address_space(3))) v4i lds_v4i;
// builtin arg0: generic pointer to 16-byte int vector (per clang diagnostic)
#define TO_GLB(p) ((v4i*)(p))
// builtin arg1: AS(3) pointer to 16-byte int vector; generic LDS pointer's
// low 32 bits are the LDS offset (ISA 10.2 aperture rule)
#define TO_LDS(p) ((lds_v4i*)(unsigned)(uintptr_t)(p))

#if USE_ASYNC
#if __has_builtin(__builtin_amdgcn_s_wait_asynccnt)
#define WAIT_ASYNC() __builtin_amdgcn_s_wait_asynccnt(0)
#else
#define WAIT_ASYNC() asm volatile("s_wait_asynccnt 0" ::: "memory")
#endif
#endif

static __device__ inline unsigned pack2_bf16(float a, float b) {
    union { __hip_bfloat16 h[2]; unsigned u; } p;
    p.h[0] = __float2bfloat16(a);
    p.h[1] = __float2bfloat16(b);
    return p.u;
}

// ---------------------------------------------------------------------------
// Kernel 1: per-token norm / variance / sum; writes
//   xn  [b][n][c]  : normalized x, bf16 (Q/K operand)
//   xvT [b][c][n]  : raw x transposed, bf16 (V operand, feature-major)
// 1 wave per token, 8 waves per block.
// ---------------------------------------------------------------------------
__global__ __launch_bounds__(256) void prep_kernel(const float* __restrict__ x,
                                                   uint2* __restrict__ xn_u,
                                                   __hip_bfloat16* __restrict__ xvT,
                                                   float* __restrict__ diff,
                                                   float* __restrict__ tokSum) {
    int wave = threadIdx.x >> 5, lane = threadIdx.x & 31;
    int tok = blockIdx.x * 8 + wave;            // < 8192
    const float4* px = (const float4*)(x + (size_t)tok * Cdim);

    float sum = 0.f, sq = 0.f;
    float4 vals[4];
#pragma unroll
    for (int k = 0; k < 4; k++) {
        float4 v = px[lane + k * 32];
        vals[k] = v;
        sum += v.x + v.y + v.z + v.w;
        sq  += v.x * v.x + v.y * v.y + v.z * v.z + v.w * v.w;
    }
#pragma unroll
    for (int m = 16; m >= 1; m >>= 1) {
        sum += __shfl_xor(sum, m, 32);
        sq  += __shfl_xor(sq,  m, 32);
    }
    float rnorm = 1.0f / fmaxf(sqrtf(sq), 1e-12f);
    if (lane == 0) {
        float var = (sq - sum * sum * (1.0f / (float)Cdim)) * (1.0f / (float)(Cdim - 1));
        diff[tok] = var;
        tokSum[tok] = sum;
    }
    // normalized bf16 (row-major)
    uint2* dst = xn_u + (size_t)tok * (Cdim / 4);
#pragma unroll
    for (int k = 0; k < 4; k++) {
        float4 v = vals[k];
        uint2 p;
        p.x = pack2_bf16(v.x * rnorm, v.y * rnorm);
        p.y = pack2_bf16(v.z * rnorm, v.w * rnorm);
        dst[lane + k * 32] = p;
    }
    // raw bf16 transposed: index (b*Cdim + c)*Ntok + n
    size_t tbase = ((size_t)(tok >> 11) * Cdim) * Ntok + (size_t)(tok & (Ntok - 1));
#pragma unroll
    for (int k = 0; k < 4; k++) {
        int c0 = (lane + k * 32) * 4;
        float4 v = vals[k];
        xvT[tbase + (size_t)(c0 + 0) * Ntok] = __float2bfloat16(v.x);
        xvT[tbase + (size_t)(c0 + 1) * Ntok] = __float2bfloat16(v.y);
        xvT[tbase + (size_t)(c0 + 2) * Ntok] = __float2bfloat16(v.z);
        xvT[tbase + (size_t)(c0 + 3) * Ntok] = __float2bfloat16(v.w);
    }
}

// ---------------------------------------------------------------------------
// Kernel 2: global stats + per-batch gate. Single block.
// stats[0]=diff_mean, stats[1]=1/(std+1e-6), stats[2..5]=gate[b]
// ---------------------------------------------------------------------------
__global__ __launch_bounds__(256) void stats_kernel(const float* __restrict__ diff,
                                                    const float* __restrict__ tokSum,
                                                    const float* __restrict__ gate_w,
                                                    const float* __restrict__ gate_b,
                                                    float* __restrict__ stats) {
    __shared__ float sh[256];
    __shared__ float res[6];
    int t = threadIdx.x;
    float d = 0.f, d2 = 0.f, bsum[4] = {0.f, 0.f, 0.f, 0.f};
    for (int i = t; i < NTOK_TOTAL; i += 256) {
        float v = diff[i];
        d += v; d2 += v * v;
        bsum[i >> 11] += tokSum[i];
    }
    float vals[6] = {d, d2, bsum[0], bsum[1], bsum[2], bsum[3]};
    for (int k = 0; k < 6; k++) {
        sh[t] = vals[k];
        __syncthreads();
        for (int off = 128; off >= 1; off >>= 1) {
            if (t < off) sh[t] += sh[t + off];
            __syncthreads();
        }
        if (t == 0) res[k] = sh[0];
        __syncthreads();
    }
    if (t == 0) {
        float mean = res[0] / (float)NTOK_TOTAL;
        float var  = (res[1] - res[0] * res[0] / (float)NTOK_TOTAL) / (float)(NTOK_TOTAL - 1);
        float stdv = sqrtf(fmaxf(var, 0.0f));
        stats[0] = mean;
        stats[1] = 1.0f / (stdv + 1e-6f);
        float w = gate_w[0], bb = gate_b[0];
        for (int b = 0; b < 4; b++) {
            float gi = res[2 + b] / (float)(Ntok * Cdim);
            stats[2 + b] = 1.0f / (1.0f + __expf(-(gi * w + bb)));
        }
    }
}

// ---------------------------------------------------------------------------
// Kernel 3: per-token combined scalar coef = gate[b]*(1+0.5*edge_mask)
// ---------------------------------------------------------------------------
__global__ __launch_bounds__(256) void coef_kernel(const float* __restrict__ diff,
                                                   const float* __restrict__ stats,
                                                   float* __restrict__ coef) {
    int i = blockIdx.x * 256 + threadIdx.x;
    float em = 1.0f / (1.0f + __expf(-(diff[i] - stats[0]) * stats[1]));
    coef[i] = stats[2 + (i >> 11)] * (1.0f + 0.5f * em);
}

// ---------------------------------------------------------------------------
// Stage one K tile (64 tok x 128 feat, token-major) and one V tile
// (128 feat x 64 tok, feature-major). Pure 16B moves; async on gfx1250.
// ---------------------------------------------------------------------------
static __device__ inline void stage_tiles(int tid, int b, int h, int tok0,
                                          const __hip_bfloat16* __restrict__ xn,
                                          const __hip_bfloat16* __restrict__ xvT,
                                          __hip_bfloat16* Kbuf, __hip_bfloat16* Vbuf) {
    // K: thread -> (row of 64, 32-feat chunk of 4)
    {
        int row = tid >> 2, part = tid & 3;
        const __hip_bfloat16* src = xn + ((size_t)b * Ntok + tok0 + row) * Cdim + h * HD + part * 32;
        __hip_bfloat16* dst = Kbuf + row * HD + part * 32;
#if USE_ASYNC
#pragma unroll
        for (int i = 0; i < 4; i++)
            __builtin_amdgcn_global_load_async_to_lds_b128(TO_GLB(src + i * 8), TO_LDS(dst + i * 8), 0, 0);
#else
        const uint4* s4 = (const uint4*)src;
        uint4* d4 = (uint4*)dst;
#pragma unroll
        for (int i = 0; i < 4; i++) d4[i] = s4[i];
#endif
    }
    // V: thread -> (feature row of 128, 32-token half)
    {
        int frow = tid >> 1, half = tid & 1;
        const __hip_bfloat16* src = xvT + ((size_t)b * Cdim + h * HD + frow) * Ntok + tok0 + half * 32;
        __hip_bfloat16* dst = Vbuf + frow * BN + half * 32;
#if USE_ASYNC
#pragma unroll
        for (int i = 0; i < 4; i++)
            __builtin_amdgcn_global_load_async_to_lds_b128(TO_GLB(src + i * 8), TO_LDS(dst + i * 8), 0, 0);
#else
        const uint4* s4 = (const uint4*)src;
        uint4* d4 = (uint4*)dst;
#pragma unroll
        for (int i = 0; i < 4; i++) d4[i] = s4[i];
#endif
    }
}

// ---------------------------------------------------------------------------
// Kernel 4: flash attention per (b, h, 128-row block). 8 waves, 16 rows/wave.
// Double-buffered LDS staging overlapped with WMMA compute.
// ---------------------------------------------------------------------------
__global__ __launch_bounds__(256) void attn_kernel(const __hip_bfloat16* __restrict__ xn,
                                                   const __hip_bfloat16* __restrict__ xvT,
                                                   const float* __restrict__ x,
                                                   const float* __restrict__ coef,
                                                   float* __restrict__ out) {
    int bid = blockIdx.x;                 // ((b*Hh + h)*16 + rb)
    int rb = bid & 15;
    int h  = (bid >> 4) & 3;
    int b  = bid >> 6;
    int tid = threadIdx.x;
    int wave = tid >> 5, lane = tid & 31;

    __shared__ __align__(16) __hip_bfloat16 KtS[2][BN * HD];   // [tok][feat]
    __shared__ __align__(16) __hip_bfloat16 VtS[2][HD * BN];   // [feat][tok]
    __shared__ __align__(16) __hip_bfloat16 Ps[8][16 * BN];    // per-wave P strip

    const int n0 = rb * 128 + wave * 16;
    const __hip_bfloat16* qbase = xn + (size_t)b * Ntok * Cdim + h * HD;

    // Q A-fragments: 16 rows x 128 feats, 4 k-chunks of 32
    v16bf qf[4];
    {
        int row = n0 + (lane & 15);
        int sel = (lane < 16) ? 0 : 8;
        const __hip_bfloat16* qrow = qbase + (size_t)row * Cdim;
#pragma unroll
        for (int kc = 0; kc < 4; kc++) {
            FragU f;
            f.u[0] = *(const uint4*)(qrow + kc * 32 + sel);
            f.u[1] = *(const uint4*)(qrow + kc * 32 + 16 + sel);
            qf[kc] = f.v;
        }
    }

    v8f O[8];
#pragma unroll
    for (int j = 0; j < 8; j++) { v8f z = {}; O[j] = z; }
    float mrow[8], lrow[8];
#pragma unroll
    for (int v = 0; v < 8; v++) { mrow[v] = -INFINITY; lrow[v] = 0.0f; }

    const float scale = 0.08838834764831845f;   // 1/sqrt(128)

    // prologue: stage tile 0
    stage_tiles(tid, b, h, 0, xn, xvT, KtS[0], VtS[0]);
#if USE_ASYNC
    WAIT_ASYNC();
#endif
    __syncthreads();

    for (int it = 0; it < NIT; ++it) {
        int cur = it & 1;
        // prefetch next tile into the other buffer (overlaps WMMA below)
        if (it + 1 < NIT)
            stage_tiles(tid, b, h, (it + 1) * BN, xn, xvT, KtS[cur ^ 1], VtS[cur ^ 1]);

        const __hip_bfloat16* Kt = KtS[cur];
        const __hip_bfloat16* Vt = VtS[cur];

        // --- S = Q K^T (scaled): 4 column subtiles x 4 K-chunks ---
        v8f S[4];
#pragma unroll
        for (int ct = 0; ct < 4; ++ct) {
            v8f acc = {};
#pragma unroll
            for (int kc = 0; kc < 4; kc++) {
                FragU kf;
                int tok = ct * 16 + (lane & 15);
                int d = kc * 32 + ((lane < 16) ? 0 : 8);
                const __hip_bfloat16* p = Kt + tok * HD + d;
                kf.u[0] = *(const uint4*)p;
                kf.u[1] = *(const uint4*)(p + 16);
                acc = __builtin_amdgcn_wmma_f32_16x16x32_bf16(
                        false, qf[kc], false, kf.v, (short)0, acc, false, false);
            }
#pragma unroll
            for (int v = 0; v < 8; v++) acc[v] *= scale;
            S[ct] = acc;
        }

        // --- online softmax (row stats via 16-lane-half butterflies) ---
        float mnew[8];
#pragma unroll
        for (int v = 0; v < 8; v++) {
            float t = fmaxf(fmaxf(S[0][v], S[1][v]), fmaxf(S[2][v], S[3][v]));
            t = fmaxf(t, __shfl_xor(t, 1, 32));
            t = fmaxf(t, __shfl_xor(t, 2, 32));
            t = fmaxf(t, __shfl_xor(t, 4, 32));
            t = fmaxf(t, __shfl_xor(t, 8, 32));
            mnew[v] = fmaxf(mrow[v], t);
        }
        float rsum[8];
#pragma unroll
        for (int v = 0; v < 8; v++) {
            float s = 0.f;
#pragma unroll
            for (int ct = 0; ct < 4; ct++) {
                float p = __expf(S[ct][v] - mnew[v]);
                S[ct][v] = p;
                s += p;
            }
            s += __shfl_xor(s, 1, 32);
            s += __shfl_xor(s, 2, 32);
            s += __shfl_xor(s, 4, 32);
            s += __shfl_xor(s, 8, 32);
            rsum[v] = s;
        }
#pragma unroll
        for (int v = 0; v < 8; v++) {
            float alpha = __expf(mrow[v] - mnew[v]);
            lrow[v] = lrow[v] * alpha + rsum[v];
            mrow[v] = mnew[v];
#pragma unroll
            for (int j = 0; j < 8; j++) O[j][v] *= alpha;
        }

        // --- re-layout P (C/D layout -> A layout) via wave-private LDS strip ---
        __hip_bfloat16* pw = &Ps[wave][0];
        {
            int col = lane & 15;
            int rbase = (lane < 16) ? 0 : 8;
#pragma unroll
            for (int ct = 0; ct < 4; ct++) {
#pragma unroll
                for (int v = 0; v < 8; v++) {
                    pw[(rbase + v) * BN + ct * 16 + col] = __float2bfloat16(S[ct][v]);
                }
            }
        }
        // same-wave DS ordering guarantees visibility; no workgroup barrier needed

        // --- O += P V: 2 K-chunks x 8 output feature subtiles ---
#pragma unroll
        for (int kc2 = 0; kc2 < 2; kc2++) {
            FragU pf;
            {
                int row = lane & 15;
                int sel = (lane < 16) ? 0 : 8;
                const __hip_bfloat16* p = pw + row * BN + kc2 * 32 + sel;
                pf.u[0] = *(const uint4*)p;
                pf.u[1] = *(const uint4*)(p + 16);
            }
#pragma unroll
            for (int j = 0; j < 8; j++) {
                FragU vf;
                int jcol = j * 16 + (lane & 15);
                int n = kc2 * 32 + ((lane < 16) ? 0 : 8);
                const __hip_bfloat16* p = Vt + jcol * BN + n;
                vf.u[0] = *(const uint4*)p;
                vf.u[1] = *(const uint4*)(p + 16);
                O[j] = __builtin_amdgcn_wmma_f32_16x16x32_bf16(
                         false, pf.v, false, vf.v, (short)0, O[j], false, false);
            }
        }

#if USE_ASYNC
        WAIT_ASYNC();        // next tile landed in LDS
#endif
        __syncthreads();     // all waves done reading cur, next buffer ready
    }

    // --- epilogue: out = x + coef[row] * O / l ---
    {
        int colL = lane & 15;
        int rbase = (lane < 16) ? 0 : 8;
#pragma unroll
        for (int v = 0; v < 8; v++) {
            int row = n0 + rbase + v;
            float cf = coef[b * Ntok + row];
            float rl = 1.0f / lrow[v];
            size_t base = ((size_t)b * Ntok + row) * Cdim + h * HD;
#pragma unroll
            for (int j = 0; j < 8; j++) {
                size_t idx = base + j * 16 + colL;
                out[idx] = x[idx] + cf * (O[j][v] * rl);
            }
        }
    }
}

// ---------------------------------------------------------------------------
extern "C" void kernel_launch(void* const* d_in, const int* in_sizes, int n_in,
                              void* d_out, int out_size, void* d_ws, size_t ws_size,
                              hipStream_t stream) {
    const float* x      = (const float*)d_in[0];
    const float* gate_w = (const float*)d_in[1];
    const float* gate_b = (const float*)d_in[2];
    float* out = (float*)d_out;

    // workspace layout
    char* ws = (char*)d_ws;
    __hip_bfloat16* xn  = (__hip_bfloat16*)ws;                         // 8 MiB
    size_t off = (size_t)Bsz * Ntok * Cdim * sizeof(__hip_bfloat16);
    __hip_bfloat16* xvT = (__hip_bfloat16*)(ws + off);                 // 8 MiB
    off += (size_t)Bsz * Ntok * Cdim * sizeof(__hip_bfloat16);
    float* diff   = (float*)(ws + off);            off += NTOK_TOTAL * sizeof(float);
    float* tokSum = (float*)(ws + off);            off += NTOK_TOTAL * sizeof(float);
    float* stats  = (float*)(ws + off);            off += 64;
    float* coefB  = (float*)(ws + off);            off += NTOK_TOTAL * sizeof(float);
    (void)ws_size; (void)in_sizes; (void)n_in; (void)out_size;

    prep_kernel <<<NTOK_TOTAL / 8, 256, 0, stream>>>(x, (uint2*)xn, xvT, diff, tokSum);
    stats_kernel<<<1, 256, 0, stream>>>(diff, tokSum, gate_w, gate_b, stats);
    coef_kernel <<<NTOK_TOTAL / 256, 256, 0, stream>>>(diff, stats, coefB);
    attn_kernel <<<Bsz * Hh * (Ntok / 128), 256, 0, stream>>>(xn, xvT, x, coefB, out);
}
